// MRC_2963527434982
// MI455X (gfx1250) — matrix operand order repaired and checked by reading, not compiled, not run on hardware
//
#include <hip/hip_runtime.h>

typedef __attribute__((ext_vector_type(16))) _Float16 v16h;
typedef __attribute__((ext_vector_type(8)))  _Float16 v8h;
typedef __attribute__((ext_vector_type(8)))  float    v8f;
typedef __attribute__((ext_vector_type(4)))  int      v4i;
typedef __attribute__((ext_vector_type(4)))  float    v4f;

#define B_ 8
#define L_ 16
#define N_ 512
#define H_ 768
#define ROWS (B_*L_*N_)            /* 65536 */
#define SPAN_TOT (ROWS*N_)         /* 33554432 */
#define K2_BLOCKS 256
#define K3_BLOCKS 8192
#define WPAD 8                      /* LDS row padding (halves) -> bank-conflict-free b128 */

__device__ __forceinline__ float sigm(float z) { return 1.0f / (1.0f + __expf(-z)); }

// ALPHA=0.5, GAMMA=2, EPS=1e-7
__device__ __forceinline__ float focalf(int y, float p) {
  if (y == 1) { float q = 1.0f - p; return -0.5f * q * q * __logf(p + 1e-7f); }
  return -0.5f * p * p * __logf(1.0f - p + 1e-7f);
}

// Deterministic block reduction (256 threads = 8 waves, wave32). Result valid on thread 0.
__device__ __forceinline__ float block_sum(float v, float* sm) {
  #pragma unroll
  for (int o = 16; o > 0; o >>= 1) v += __shfl_down(v, o, 32);
  __syncthreads();
  if ((threadIdx.x & 31) == 0) sm[threadIdx.x >> 5] = v;
  __syncthreads();
  float r = (threadIdx.x < 8) ? sm[threadIdx.x] : 0.0f;
  if ((threadIdx.x >> 5) == 0) {
    #pragma unroll
    for (int o = 4; o > 0; o >>= 1) r += __shfl_down(r, o, 32);
  }
  return r;
}

// ---------------------------------------------------------------------------
// k1: fused projection GEMM  x(ROWS x H) @ Bpad(H x 16)  via v_wmma_f32_16x16x32_f16
//   Bpad col0 = W_se[:,0], col1 = W_se[:,1], col2 = W_ys[:,0], cols 3..15 = 0
// LDS holds Bpad n-major (wl[n][k], padded) so each WMMA needs only 2 ds_load_b128/lane.
// ---------------------------------------------------------------------------
__global__ __launch_bounds__(256) void k1_proj(const float* __restrict__ x,
                                               const float* __restrict__ Wse,
                                               const float* __restrict__ Wys,
                                               float* __restrict__ so0,
                                               float* __restrict__ so1,
                                               float* __restrict__ xxo)
{
  __shared__ _Float16 wl[16 * (H_ + WPAD)];   // ~24.3 KB, n-major
  for (int t = threadIdx.x; t < 16 * H_; t += 256) {
    int k = t % H_, n = t / H_;
    float w = 0.0f;
    if (n == 0)      w = Wse[2 * k];
    else if (n == 1) w = Wse[2 * k + 1];
    else if (n == 2) w = Wys[k];
    wl[n * (H_ + WPAD) + k] = (_Float16)w;
  }
  __syncthreads();

  const int lane = threadIdx.x & 31;
  const int wave = threadIdx.x >> 5;
  const int tile = blockIdx.x * 8 + wave;     // 16-row tile
  const int rowbase = tile * 16;
  const int m    = lane & 15;
  const int half = lane >> 4;
  const float* xrow = x + (size_t)(rowbase + m) * H_;
  const _Float16* brow = &wl[(lane & 15) * (H_ + WPAD) + half * 8];

  v8f c = {};
  for (int k0 = 0; k0 < H_; k0 += 32) {
    // A tile (16x32 f16): lane holds row m; VGPR0-3: K=half*8+0..7, VGPR4-7: K=16+half*8+0..7
    const float4* p0 = (const float4*)(xrow + k0 + half * 8);
    const float4* p1 = (const float4*)(xrow + k0 + 16 + half * 8);
    float4 a0 = p0[0], a1 = p0[1], a2 = p1[0], a3 = p1[1];
    v16h a;
    a[0]  = (_Float16)a0.x; a[1]  = (_Float16)a0.y; a[2]  = (_Float16)a0.z; a[3]  = (_Float16)a0.w;
    a[4]  = (_Float16)a1.x; a[5]  = (_Float16)a1.y; a[6]  = (_Float16)a1.z; a[7]  = (_Float16)a1.w;
    a[8]  = (_Float16)a2.x; a[9]  = (_Float16)a2.y; a[10] = (_Float16)a2.z; a[11] = (_Float16)a2.w;
    a[12] = (_Float16)a3.x; a[13] = (_Float16)a3.y; a[14] = (_Float16)a3.z; a[15] = (_Float16)a3.w;

    // B tile (32x16 f16): elems 0-7: K=k0+half*8+0..7, elems 8-15: K=k0+16+half*8+0..7; N = lane&15
    // n-major LDS layout -> two runs of 8 contiguous halves -> two b128 loads
    v8h b0 = *(const v8h*)(brow + k0);
    v8h b1 = *(const v8h*)(brow + k0 + 16);
    v16h b;
    #pragma unroll
    for (int e = 0; e < 8; ++e) { b[e] = b0[e]; b[8 + e] = b1[e]; }

    c = __builtin_amdgcn_wmma_f32_16x16x32_f16(false, a, false, b, (short)0, c,
                                               false, false);
  }

  // C layout: VGPR r -> M = r + 8*half, N = lane&15.  Only N<3 carries data.
  const int n = lane & 15;
  if (n < 3) {
    float* dst = (n == 0) ? so0 : (n == 1) ? so1 : xxo;
    const int base = rowbase + half * 8;
    #pragma unroll
    for (int r = 0; r < 8; ++r) dst[base + r] = c[r];
  }
}

// ---------------------------------------------------------------------------
// k2: per-(b,l,n) sigmoid, start/end preds (written to d_out), se-loss partials
// ---------------------------------------------------------------------------
__global__ __launch_bounds__(256) void k2_se(const float* __restrict__ so0,
                                             const float* __restrict__ so1,
                                             const int* __restrict__ startl,
                                             const int* __restrict__ endl,
                                             const int* __restrict__ seqlen,
                                             const float* __restrict__ bse,
                                             float* __restrict__ outS,
                                             float* __restrict__ outE,
                                             float* __restrict__ part2)
{
  __shared__ float sm[8];
  const int idx = blockIdx.x * 256 + threadIdx.x;        // < ROWS exactly
  const int b = idx / (L_ * N_);
  const int l = (idx / N_) % L_;
  const int n = idx % N_;
  const float mask = (n < seqlen[b]) ? 1.0f : 0.0f;
  const float p0 = sigm(so0[idx] + bse[0]);
  const float p1 = sigm(so1[idx] + bse[1]);
  outS[idx] = (p0 > 0.5f) ? mask : 0.0f;
  outE[idx] = (p1 > 0.5f) ? mask : 0.0f;
  const float lossv = (focalf(startl[idx], p0) + focalf(endl[idx], p1)) * mask;
  const float msum  = (l == 0) ? mask : 0.0f;            // sum(maskf) over (B,N) only
  float r0 = block_sum(lossv, sm);
  float r1 = block_sum(msum, sm);
  if (threadIdx.x == 0) { part2[blockIdx.x * 2] = r0; part2[blockIdx.x * 2 + 1] = r1; }
}

// ---------------------------------------------------------------------------
// k3: streaming pass over 33.5M (b,l,i,j), vectorized x4 in j, non-temporal
//     span/val reads and span_pred writes (pure streaming, keep L2 for xx/preds)
// ---------------------------------------------------------------------------
__global__ __launch_bounds__(256) void k3_span(const float* __restrict__ xx,
                                               const int* __restrict__ span,
                                               const int* __restrict__ val,
                                               const float* __restrict__ outS,
                                               const float* __restrict__ outE,
                                               const float* __restrict__ bys,
                                               const float* __restrict__ Wspan,
                                               const float* __restrict__ bspan,
                                               float* __restrict__ outSpan,
                                               float* __restrict__ part3)
{
  __shared__ float sm[8];
  const float w0 = Wspan[0], w1 = Wspan[1], bb = bspan[0], by = bys[0];
  float lossn = 0.0f, vals = 0.0f, tp = 0.0f, tn = 0.0f, fp = 0.0f;
  const int nquads = SPAN_TOT / 4;
  const int stride = K3_BLOCKS * 256;
  for (int q = blockIdx.x * 256 + threadIdx.x; q < nquads; q += stride) {
    const int idx = q * 4;
    const int j0  = idx & (N_ - 1);                 // 4 | N_ -> quad stays in one row
    const int i   = (idx >> 9) & (N_ - 1);
    const int bl  = idx >> 18;                      // combined (b,l)
    const float xi   = xx[bl * N_ + i] + by;
    const bool  spi  = outS[bl * N_ + i] > 0.5f;
    const v4f   xjv  = *(const v4f*)&xx[bl * N_ + j0];
    const v4f   epv  = *(const v4f*)&outE[bl * N_ + j0];
    const v4i   sv   = __builtin_nontemporal_load((const v4i*)&span[idx]);
    const v4i   vv   = __builtin_nontemporal_load((const v4i*)&val[idx]);
    v4f outv;
    #pragma unroll
    for (int e = 0; e < 4; ++e) {
      const int j = j0 + e;
      const float po = sigm(xi * w0 + (xjv[e] + by) * w1 + bb);
      const float vf = (float)vv[e];
      const bool pred = (po > 0.5f) && (i <= j) && spi && (epv[e] > 0.5f);
      outv[e] = pred ? 1.0f : 0.0f;
      lossn += focalf(sv[e], po) * vf;
      vals  += vf;
      if (sv[e] == 1) { if (pred) tp += 1.0f; else tn += 1.0f; }
      else if (pred) fp += 1.0f;
    }
    __builtin_nontemporal_store(outv, (v4f*)&outSpan[idx]);
  }
  float r;
  r = block_sum(lossn, sm); if (threadIdx.x == 0) part3[blockIdx.x * 5 + 0] = r;
  r = block_sum(vals,  sm); if (threadIdx.x == 0) part3[blockIdx.x * 5 + 1] = r;
  r = block_sum(tp,    sm); if (threadIdx.x == 0) part3[blockIdx.x * 5 + 2] = r;
  r = block_sum(tn,    sm); if (threadIdx.x == 0) part3[blockIdx.x * 5 + 3] = r;
  r = block_sum(fp,    sm); if (threadIdx.x == 0) part3[blockIdx.x * 5 + 4] = r;
}

// ---------------------------------------------------------------------------
// k4: deterministic finalize of the partial sums -> 5 scalar outputs
// ---------------------------------------------------------------------------
__global__ __launch_bounds__(256) void k4_final(const float* __restrict__ part2,
                                                const float* __restrict__ part3,
                                                float* __restrict__ scal)
{
  __shared__ float sm[8];
  float acc[7];
  for (int q = 0; q < 7; ++q) {
    float v = 0.0f;
    if (q < 2) {
      for (int i = threadIdx.x; i < K2_BLOCKS; i += 256) v += part2[i * 2 + q];
    } else {
      const int f = q - 2;
      for (int i = threadIdx.x; i < K3_BLOCKS; i += 256) v += part3[i * 5 + f];
    }
    acc[q] = block_sum(v, sm);
  }
  if (threadIdx.x == 0) {
    scal[0] = acc[4];                              // tp
    scal[1] = acc[5];                              // tn
    scal[2] = acc[6];                              // fp
    scal[3] = acc[0] / (acc[1] * (2.0f * L_));     // se_loss
    scal[4] = acc[2] / (acc[3] + 1e-7f);           // span_loss
  }
}

extern "C" void kernel_launch(void* const* d_in, const int* in_sizes, int n_in,
                              void* d_out, int out_size, void* d_ws, size_t ws_size,
                              hipStream_t stream)
{
  (void)in_sizes; (void)n_in; (void)out_size; (void)ws_size;
  const float* x      = (const float*)d_in[0];
  const int*   st     = (const int*)d_in[1];
  const int*   en     = (const int*)d_in[2];
  const int*   span   = (const int*)d_in[3];
  const int*   val    = (const int*)d_in[4];
  const int*   seqlen = (const int*)d_in[5];
  const float* Wse    = (const float*)d_in[6];
  const float* bse    = (const float*)d_in[7];
  const float* Wys    = (const float*)d_in[8];
  const float* bys    = (const float*)d_in[9];
  const float* Wspan  = (const float*)d_in[10];
  const float* bspan  = (const float*)d_in[11];
  float* out = (float*)d_out;
  float* ws  = (float*)d_ws;

  float* so0   = ws;
  float* so1   = ws + ROWS;
  float* xxw   = ws + 2 * ROWS;
  float* part2 = ws + 3 * ROWS;
  float* part3 = part2 + K2_BLOCKS * 2;

  float* outSpan = out;                 // (B,L,N,N) span_pred as float 0/1
  float* outS    = out + SPAN_TOT;      // start_pred
  float* outE    = outS + ROWS;         // end_pred
  float* scal    = outE + ROWS;         // tp, tn, fp, se_loss, span_loss

  k1_proj<<<ROWS / 128, 256, 0, stream>>>(x, Wse, Wys, so0, so1, xxw);
  k2_se<<<K2_BLOCKS, 256, 0, stream>>>(so0, so1, st, en, seqlen, bse, outS, outE, part2);
  k3_span<<<K3_BLOCKS, 256, 0, stream>>>(xxw, span, val, outS, outE, bys, Wspan, bspan,
                                         outSpan, part3);
  k4_final<<<1, 256, 0, stream>>>(part2, part3, scal);
}
